// MiniGPT_73650099192380
// MI455X (gfx1250) — matrix operand order repaired and checked by reading, not compiled, not run on hardware
//
#include <hip/hip_runtime.h>
#include <hip/hip_bf16.h>

// ---------------------------------------------------------------------------
// Types for CDNA5 WMMA (wave32)
// ---------------------------------------------------------------------------
typedef __bf16  v16bf  __attribute__((ext_vector_type(16)));
typedef __bf16  bf16x8 __attribute__((ext_vector_type(8)));
typedef float   v8f    __attribute__((ext_vector_type(8)));

union Frag {
    v16bf  v;
    bf16x8 h[2];
};

__device__ __forceinline__ unsigned short f2bf(float x) {
    unsigned int u = __float_as_uint(x);
    u += 0x7FFFu + ((u >> 16) & 1u);          // round-to-nearest-even
    return (unsigned short)(u >> 16);
}

// ---------------------------------------------------------------------------
// Embedding: x[b,t,:] = tok_emb[idx[b,t],:] + pos_emb[t,:]
// ---------------------------------------------------------------------------
__global__ __launch_bounds__(256) void embed_kernel(
    const int* __restrict__ idx, const float* __restrict__ tok_emb,
    const float* __restrict__ pos_emb, float* __restrict__ x)
{
    const int row = blockIdx.x;            // b*1024 + t
    const int t   = row & 1023;
    const int tok = idx[row];
    const int c   = threadIdx.x * 4;
    float4 te = *(const float4*)&tok_emb[(size_t)tok * 1024 + c];
    float4 pe = *(const float4*)&pos_emb[(size_t)t   * 1024 + c];
    float4 o;
    o.x = te.x + pe.x; o.y = te.y + pe.y; o.z = te.z + pe.z; o.w = te.w + pe.w;
    *(float4*)&x[(size_t)row * 1024 + c] = o;
}

// ---------------------------------------------------------------------------
// LayerNorm over D=1024. grid = 4096 rows, 256 threads (4 floats each).
// ---------------------------------------------------------------------------
__global__ __launch_bounds__(256) void ln_kernel(
    const float* __restrict__ x, const float* __restrict__ w,
    const float* __restrict__ b, float* __restrict__ out)
{
    __shared__ float ssum[256];
    __shared__ float ssq[256];
    const int row = blockIdx.x;
    const int tid = threadIdx.x;
    const int c   = tid * 4;
    float4 v = *(const float4*)&x[(size_t)row * 1024 + c];
    float s  = v.x + v.y + v.z + v.w;
    float q  = v.x * v.x + v.y * v.y + v.z * v.z + v.w * v.w;
    ssum[tid] = s; ssq[tid] = q;
    __syncthreads();
    for (int off = 128; off > 0; off >>= 1) {
        if (tid < off) { ssum[tid] += ssum[tid + off]; ssq[tid] += ssq[tid + off]; }
        __syncthreads();
    }
    const float mu   = ssum[0] * (1.0f / 1024.0f);
    const float var  = ssq[0] * (1.0f / 1024.0f) - mu * mu;
    const float rstd = rsqrtf(var + 1e-5f);
    float4 wv = *(const float4*)&w[c];
    float4 bv = *(const float4*)&b[c];
    float4 o;
    o.x = (v.x - mu) * rstd * wv.x + bv.x;
    o.y = (v.y - mu) * rstd * wv.y + bv.y;
    o.z = (v.z - mu) * rstd * wv.z + bv.z;
    o.w = (v.w - mu) * rstd * wv.w + bv.w;
    *(float4*)&out[(size_t)row * 1024 + c] = o;
}

// ---------------------------------------------------------------------------
// WMMA bf16 GEMM: C[M,N] = epilogue(A[M,K] @ W[K,N])
// Block tile 128x128x32, 256 threads = 8 waves (2 x 4), each wave 64x32.
// Software pipeline: next tile staged in registers during WMMA of current
// tile; tile after next pulled toward L2 with global_prefetch_b8.
// Epilogue is compile-time selected: + bias[n], + resid[m,n], ReLU.
// All dims are multiples of 128 (M,N) and 32 (K): no bounds checks.
// ---------------------------------------------------------------------------
#define LDK 40   // padded bf16 row length (80B: 16B-aligned, conflict-free)

template<bool HAS_BIAS, bool HAS_RES, bool DO_RELU>
__global__ __launch_bounds__(256) void gemm_kernel(
    const float* __restrict__ A, const float* __restrict__ W,
    float* __restrict__ C, int M, int N, int K,
    const float* __restrict__ bias, const float* __restrict__ resid)
{
    __shared__ unsigned short Alds[128][LDK];   // [m][k]
    __shared__ unsigned short Blds[128][LDK];   // [n][k] (transposed W tile)

    const int tid  = threadIdx.x;
    const int lane = tid & 31;
    const int wid  = tid >> 5;          // 0..7
    const int wm   = wid >> 2;          // 0..1  (64-row stripe)
    const int wn   = wid & 3;           // 0..3  (32-col stripe)
    const int lr   = lane & 15;
    const int half = lane >> 4;

    const int m0 = blockIdx.y * 128;
    const int n0 = blockIdx.x * 128;

    // Per-thread global staging coordinates (fixed across k-steps)
    const int arow = tid >> 3;          // 0..31 (+ p*32)
    const int acol = (tid & 7) << 2;    // 0,4,..,28
    const int wrow = tid >> 5;          // 0..7  (+ p*8)
    const int wcol = (tid & 31) << 2;   // 0,4,..,124

    v8f acc[4][2];
#pragma unroll
    for (int tm = 0; tm < 4; ++tm)
#pragma unroll
        for (int tn = 0; tn < 2; ++tn)
            acc[tm][tn] = (v8f)0.0f;

    float4 aS[4], wS[4];                // register-staged next tile

    auto gload = [&](int k0) {
#pragma unroll
        for (int p = 0; p < 4; ++p)
            aS[p] = *(const float4*)&A[(size_t)(m0 + arow + p * 32) * K + k0 + acol];
#pragma unroll
        for (int p = 0; p < 4; ++p)
            wS[p] = *(const float4*)&W[(size_t)(k0 + wrow + p * 8) * N + n0 + wcol];
    };

    auto stage = [&]() {
#pragma unroll
        for (int p = 0; p < 4; ++p) {
            unsigned short* d = &Alds[arow + p * 32][acol];
            d[0] = f2bf(aS[p].x); d[1] = f2bf(aS[p].y);
            d[2] = f2bf(aS[p].z); d[3] = f2bf(aS[p].w);
        }
#pragma unroll
        for (int p = 0; p < 4; ++p) {
            const int kk = wrow + p * 8;
            Blds[wcol + 0][kk] = f2bf(wS[p].x);
            Blds[wcol + 1][kk] = f2bf(wS[p].y);
            Blds[wcol + 2][kk] = f2bf(wS[p].z);
            Blds[wcol + 3][kk] = f2bf(wS[p].w);
        }
    };

    auto compute = [&]() {
        Frag fa[4], fb[2];
#pragma unroll
        for (int tm = 0; tm < 4; ++tm) {
            const unsigned short* p = &Alds[wm * 64 + tm * 16 + lr][0];
            // A layout (ISA 7.12.2): half 0 -> K {0..7,16..23}, half 1 -> K {8..15,24..31}
            fa[tm].h[0] = *(const bf16x8*)(p + half * 8);
            fa[tm].h[1] = *(const bf16x8*)(p + 16 + half * 8);
        }
#pragma unroll
        for (int tn = 0; tn < 2; ++tn) {
            const unsigned short* p = &Blds[wn * 32 + tn * 16 + lr][0];
            // B layout: half 0 -> K {0..7,16..23}, half 1 -> K {8..15,24..31}
            fb[tn].h[0] = *(const bf16x8*)(p + half * 8);
            fb[tn].h[1] = *(const bf16x8*)(p + 16 + half * 8);
        }
#pragma unroll
        for (int tm = 0; tm < 4; ++tm)
#pragma unroll
            for (int tn = 0; tn < 2; ++tn)
                acc[tm][tn] = __builtin_amdgcn_wmma_f32_16x16x32_bf16(
                    false, fa[tm].v, false, fb[tn].v,
                    (short)0, acc[tm][tn], false, false);
    };

    // ---- software-pipelined K loop ----
    gload(0);
    stage();
    __syncthreads();
    const int steps = K >> 5;
    for (int it = 1; it < steps; ++it) {
        const int k0 = it << 5;
        gload(k0);                                   // next tile -> regs
        if (it + 1 < steps) {                        // tile after next -> cache
            const int kp = (it + 1) << 5;
            __builtin_prefetch(&A[(size_t)(m0 + arow) * K + kp + acol], 0, 3);
            __builtin_prefetch(&W[(size_t)(kp + wrow) * N + n0 + wcol], 0, 3);
        }
        compute();                                   // WMMAs on current LDS tile
        __syncthreads();                             // all waves done reading
        stage();                                     // regs -> LDS
        __syncthreads();
    }
    compute();                                       // last tile

    // ---- straight-line epilogue (compile-time variant) ----
#pragma unroll
    for (int tm = 0; tm < 4; ++tm) {
#pragma unroll
        for (int tn = 0; tn < 2; ++tn) {
            const int gm0 = m0 + wm * 64 + tm * 16 + 8 * half;
            const int gn  = n0 + wn * 32 + tn * 16 + lr;
            float bi = 0.0f;
            if (HAS_BIAS) bi = bias[gn];
#pragma unroll
            for (int r = 0; r < 8; ++r) {
                const int gm = gm0 + r;
                float vv = acc[tm][tn][r];
                if (HAS_BIAS) vv += bi;
                if (HAS_RES)  vv += resid[(size_t)gm * N + gn];
                if (DO_RELU)  vv = fmaxf(vv, 0.0f);
                C[(size_t)gm * N + gn] = vv;
            }
        }
    }
}

// ---------------------------------------------------------------------------
// Causal attention, one thread per query row, online softmax, K/V tiles in LDS.
// grid = (T/256, H, B), block = 256.  HD = 64, scale = 1/8.
// ---------------------------------------------------------------------------
__global__ __launch_bounds__(256) void attn_kernel(
    const float* __restrict__ q, const float* __restrict__ k,
    const float* __restrict__ v, float* __restrict__ out)
{
    __shared__ float Ksh[64][64];
    __shared__ float Vsh[64][64];
    const int tid = threadIdx.x;
    const int t   = blockIdx.x * 256 + tid;
    const int h   = blockIdx.y;
    const int b   = blockIdx.z;
    const size_t rowbase = ((size_t)b * 1024 + t) * 1024 + h * 64;

    float qv[64], acc[64];
#pragma unroll
    for (int j = 0; j < 64; ++j) { qv[j] = q[rowbase + j]; acc[j] = 0.0f; }

    float mrun = -1e30f, lrun = 0.0f;
    const int ntiles = blockIdx.x * 4 + 4;     // key tiles of 64 up to block max t

    for (int kt = 0; kt < ntiles; ++kt) {
        // cooperative K/V tile load (64x64 f32 each)
#pragma unroll
        for (int p = 0; p < 4; ++p) {
            int i  = p * 256 + tid;
            int r  = i >> 4;
            int c4 = (i & 15) << 2;
            size_t g = ((size_t)b * 1024 + kt * 64 + r) * 1024 + h * 64 + c4;
            *(float4*)&Ksh[r][c4] = *(const float4*)&k[g];
            *(float4*)&Vsh[r][c4] = *(const float4*)&v[g];
        }
        __syncthreads();

        const int kmax = t - kt * 64;                    // causal bound in tile
        const int kend = (kmax >= 63) ? 64 : (kmax + 1); // may be <= 0
        for (int kk = 0; kk < kend; ++kk) {
            float s = 0.0f;
#pragma unroll 8
            for (int j = 0; j < 64; ++j) s += qv[j] * Ksh[kk][j];
            s *= 0.125f;
            if (s > mrun) {
                const float corr = __expf(mrun - s);
                lrun *= corr;
#pragma unroll 8
                for (int j = 0; j < 64; ++j) acc[j] *= corr;
                mrun = s;
            }
            const float p = __expf(s - mrun);
            lrun += p;
#pragma unroll 8
            for (int j = 0; j < 64; ++j) acc[j] += p * Vsh[kk][j];
        }
        __syncthreads();
    }

    const float inv = 1.0f / lrun;
#pragma unroll
    for (int j = 0; j < 64; ++j) out[rowbase + j] = acc[j] * inv;
}

// ---------------------------------------------------------------------------
// Orchestration
// ---------------------------------------------------------------------------
extern "C" void kernel_launch(void* const* d_in, const int* in_sizes, int n_in,
                              void* d_out, int out_size, void* d_ws, size_t ws_size,
                              hipStream_t stream)
{
    (void)in_sizes; (void)n_in; (void)out_size; (void)ws_size;

    const int*   idx     = (const int*)  d_in[0];
    const float* tok_emb = (const float*)d_in[1];
    const float* pos_emb = (const float*)d_in[2];
    const float* ln1_w   = (const float*)d_in[3];
    const float* ln1_b   = (const float*)d_in[4];
    const float* wq      = (const float*)d_in[5];
    const float* wk      = (const float*)d_in[6];
    const float* wvw     = (const float*)d_in[7];
    const float* wproj   = (const float*)d_in[8];
    const float* bproj   = (const float*)d_in[9];
    const float* ln2_w   = (const float*)d_in[10];
    const float* ln2_b   = (const float*)d_in[11];
    const float* w1      = (const float*)d_in[12];
    const float* b1      = (const float*)d_in[13];
    const float* w2      = (const float*)d_in[14];
    const float* b2      = (const float*)d_in[15];
    const float* lnf_w   = (const float*)d_in[16];
    const float* lnf_b   = (const float*)d_in[17];
    const float* wout    = (const float*)d_in[18];
    const float* bout    = (const float*)d_in[19];

    // Workspace layout (floats): x[4M] | a[4M] | ff[16M] (ff reused for q/k/v)
    float* ws = (float*)d_ws;
    const size_t MT = (size_t)4096 * 1024;     // 4M floats
    float* x  = ws;
    float* a  = ws + MT;
    float* ff = ws + 2 * MT;
    float* qb = ff;
    float* kb = ff + MT;
    float* vb = ff + 2 * MT;

    const dim3 blk(256);
    const dim3 gD(1024 / 128, 4096 / 128);     // M=4096, N=1024
    const dim3 gFF(4096 / 128, 4096 / 128);    // M=4096, N=4096
    const dim3 gOut(32000 / 128, 4096 / 128);  // M=4096, N=32000
    const dim3 gAttn(4, 16, 4);                // (T/256, H, B)

    embed_kernel<<<4096, blk, 0, stream>>>(idx, tok_emb, pos_emb, x);

    for (int l = 0; l < 4; ++l) {
        const size_t oDD = (size_t)l * 1024 * 1024;
        const size_t oDF = (size_t)l * 1024 * 4096;

        ln_kernel<<<4096, blk, 0, stream>>>(x, ln1_w + l * 1024, ln1_b + l * 1024, a);
        gemm_kernel<false, false, false><<<gD, blk, 0, stream>>>(
            a, wq  + oDD, qb, 4096, 1024, 1024, nullptr, nullptr);
        gemm_kernel<false, false, false><<<gD, blk, 0, stream>>>(
            a, wk  + oDD, kb, 4096, 1024, 1024, nullptr, nullptr);
        gemm_kernel<false, false, false><<<gD, blk, 0, stream>>>(
            a, wvw + oDD, vb, 4096, 1024, 1024, nullptr, nullptr);

        attn_kernel<<<gAttn, blk, 0, stream>>>(qb, kb, vb, a);

        gemm_kernel<true, true, false><<<gD, blk, 0, stream>>>(
            a, wproj + oDD, x, 4096, 1024, 1024, bproj + l * 1024, x);

        ln_kernel<<<4096, blk, 0, stream>>>(x, ln2_w + l * 1024, ln2_b + l * 1024, a);
        gemm_kernel<true, false, true><<<gFF, blk, 0, stream>>>(
            a, w1 + oDF, ff, 4096, 4096, 1024, b1 + l * 4096, nullptr);
        gemm_kernel<true, true, false><<<gD, blk, 0, stream>>>(
            ff, w2 + oDF, x, 4096, 1024, 4096, b2 + l * 1024, x);
    }

    ln_kernel<<<4096, blk, 0, stream>>>(x, lnf_w, lnf_b, a);
    gemm_kernel<true, false, false><<<gOut, blk, 0, stream>>>(
        a, wout, (float*)d_out, 4096, 32000, 1024, bout, nullptr);
}
